// Block_33191507264060
// MI455X (gfx1250) — matrix operand order repaired and checked by reading, not compiled, not run on hardware
//
#include <hip/hip_runtime.h>

// ---------------------------------------------------------------------------
// Transformer block for MI455X (gfx1250, wave32, WMMA).
// All GEMM-like contractions use v_wmma_f32_16x16x32_f16 (f16 in, f32 acc).
// GEMMs use 32x32 register tiles per wave: 2 A-frags x 2 B-frags -> 4
// independent WMMA accumulator chains (keeps the matrix pipe full, halves
// L2 traffic per WMMA). Attention is flash-style: scores never touch HBM;
// the P tile is re-laid-out C-frag -> A-frag through a 1KB/wave LDS buffer.
// ---------------------------------------------------------------------------

typedef __attribute__((ext_vector_type(16))) _Float16 v16h;
typedef __attribute__((ext_vector_type(8)))  _Float16 v8h;
typedef __attribute__((ext_vector_type(8)))  float    v8f;

constexpr int Bn = 8, Tn = 1024, Cn = 384, Hn = 6, Dn = 64;
constexpr int Mn = Bn * Tn;        // 8192 rows
constexpr int Fn = 4 * Cn;         // 1536

// ---- WMMA fragment helpers (wave32 layouts per CDNA5 ISA 7.12.2) ----------
// A 16x32 f16: lanes 0..15: M=lane, K-half base 0; lanes 16..31: M=lane-16,
// K-half base 8. elems 0..7 -> K=ko+0..7 ; elems 8..15 -> K=16+ko+0..7.
__device__ __forceinline__ v16h ldA16(const _Float16* base, int ld) {
  int lane = threadIdx.x & 31;
  int m  = lane & 15;
  int ko = (lane & 16) ? 8 : 0;
  const _Float16* p = base + (size_t)m * ld + ko;
  v8h lo = *(const v8h*)(p);
  v8h hi = *(const v8h*)(p + 16);
  return __builtin_shufflevector(lo, hi, 0,1,2,3,4,5,6,7,8,9,10,11,12,13,14,15);
}
// B 32x16 f16 from Bt stored [N][K] row-major: lanes 0..15 N=lane, K-half base 0;
// lanes 16..31 N=lane-16, K-half base 8. elems 0..7 -> K=ko+0..7 ; elems 8..15 -> K=16+ko+0..7.
__device__ __forceinline__ v16h ldB16(const _Float16* base, int ld) {
  int lane = threadIdx.x & 31;
  int n  = lane & 15;
  int ko = (lane & 16) ? 8 : 0;
  const _Float16* p = base + (size_t)n * ld + ko;
  v8h lo = *(const v8h*)(p);
  v8h hi = *(const v8h*)(p + 16);
  return __builtin_shufflevector(lo, hi, 0,1,2,3,4,5,6,7,8,9,10,11,12,13,14,15);
}
__device__ __forceinline__ v8f wmma16(v16h a, v16h b, v8f c) {
  return __builtin_amdgcn_wmma_f32_16x16x32_f16(false, a, false, b,
                                                (short)0, c, false, false);
}

// ---- weight transpose + fp32 -> f16 cast: in[b][r][c] -> out[b][c][r] -----
__global__ void transpose_cvt_kernel(const float* __restrict__ in,
                                     _Float16* __restrict__ out,
                                     int Bt, int R, int Cc) {
  size_t total = (size_t)Bt * R * Cc;
  for (size_t i = (size_t)blockIdx.x * blockDim.x + threadIdx.x; i < total;
       i += (size_t)gridDim.x * blockDim.x) {
    int c = (int)(i % Cc);
    size_t t = i / Cc;
    int r = (int)(t % R);
    int b = (int)(t / R);
    out[((size_t)b * Cc + c) * R + r] = (_Float16)in[i];
  }
}

// ---- LayerNorm over C=384, one wave per row, f16 output -------------------
__global__ void layernorm_kernel(const float* __restrict__ x,
                                 const float* __restrict__ g,
                                 const float* __restrict__ be,
                                 _Float16* __restrict__ out) {
  int row  = blockIdx.x * 4 + threadIdx.y;
  int lane = threadIdx.x;
  const float* xr = x + (size_t)row * Cn;
  float v[12];
  float s = 0.f;
#pragma unroll
  for (int i = 0; i < 12; i++) { v[i] = xr[lane + 32 * i]; s += v[i]; }
#pragma unroll
  for (int m = 1; m < 32; m <<= 1) s += __shfl_xor(s, m, 32);
  float mu = s * (1.f / Cn);
  float var = 0.f;
#pragma unroll
  for (int i = 0; i < 12; i++) { float d = v[i] - mu; var += d * d; }
#pragma unroll
  for (int m = 1; m < 32; m <<= 1) var += __shfl_xor(var, m, 32);
  float rs = rsqrtf(var * (1.f / Cn) + 1e-5f);
  _Float16* o = out + (size_t)row * Cn;
#pragma unroll
  for (int i = 0; i < 12; i++) {
    int c = lane + 32 * i;
    o[c] = (_Float16)((v[i] - mu) * rs * g[c] + be[c]);
  }
}

// ---- fused QKV projection, 32x32 tiles; V stored transposed [b][h][d][t] --
__global__ void qkv_kernel(const _Float16* __restrict__ h1,
                           const _Float16* __restrict__ wT,   // [3][H][D][C]
                           _Float16* __restrict__ q,
                           _Float16* __restrict__ k,
                           _Float16* __restrict__ vT) {
  int n0 = blockIdx.x * 32;                 // 0..1151; never crosses head/which
  int m0 = (blockIdx.y * 4 + threadIdx.y) * 32;
  int which = n0 / Cn;
  int rem   = n0 - which * Cn;
  int head  = rem / Dn;
  int d0    = rem - head * Dn;              // 0 or 32
  const _Float16* B0 = wT + (((size_t)which * Hn + head) * Dn + d0) * Cn;
  const _Float16* B1 = B0 + (size_t)16 * Cn;
  const _Float16* A0 = h1 + (size_t)m0 * Cn;
  const _Float16* A1 = A0 + (size_t)16 * Cn;
  v8f c00 = {}, c01 = {}, c10 = {}, c11 = {};
  for (int kk = 0; kk < Cn; kk += 32) {
    v16h a0 = ldA16(A0 + kk, Cn), a1 = ldA16(A1 + kk, Cn);
    v16h b0 = ldB16(B0 + kk, Cn), b1 = ldB16(B1 + kk, Cn);
    c00 = wmma16(a0, b0, c00); c01 = wmma16(a0, b1, c01);
    c10 = wmma16(a1, b0, c10); c11 = wmma16(a1, b1, c11);
  }
  int lane = threadIdx.x;
  int nc = lane & 15;
  int rb = (lane & 16) ? 8 : 0;
#pragma unroll
  for (int mi = 0; mi < 2; mi++) {
#pragma unroll
    for (int ni = 0; ni < 2; ni++) {
      v8f acc = (mi == 0) ? (ni == 0 ? c00 : c01) : (ni == 0 ? c10 : c11);
      int d = d0 + ni * 16 + nc;
#pragma unroll
      for (int r = 0; r < 8; r++) {
        int m = m0 + mi * 16 + rb + r;
        int b = m >> 10, t = m & 1023;
        _Float16 val = (_Float16)acc[r];
        if (which == 0)
          q[(((size_t)b * Hn + head) * Tn + t) * Dn + d] = val;
        else if (which == 1)
          k[(((size_t)b * Hn + head) * Tn + t) * Dn + d] = val;
        else
          vT[(((size_t)b * Hn + head) * Dn + d) * Tn + t] = val;
      }
    }
  }
}

// ---- flash attention: one wave owns 16 query rows of one (b,h) ------------
__global__ void attn_kernel(const _Float16* __restrict__ q,
                            const _Float16* __restrict__ k,
                            const _Float16* __restrict__ vT,
                            _Float16* __restrict__ out) {
  __shared__ __align__(32) _Float16 pbuf[4][16 * 32];  // P staging, 1KB/wave
  int bh   = blockIdx.y;                 // b*H + h
  int qt   = blockIdx.x * 4 + threadIdx.y;
  int lane = threadIdx.x;
  _Float16* myp = pbuf[threadIdx.y];
  const _Float16* qb  = q  + ((size_t)bh * Tn + qt * 16) * Dn;
  const _Float16* kb0 = k  + (size_t)bh * Tn * Dn;
  const _Float16* vb0 = vT + (size_t)bh * Dn * Tn;
  v16h qa0 = ldA16(qb, Dn);              // q tile K=0..31
  v16h qa1 = ldA16(qb + 32, Dn);         // q tile K=32..63
  float mrow[8], lrow[8];
#pragma unroll
  for (int r = 0; r < 8; r++) { mrow[r] = -1e30f; lrow[r] = 0.f; }
  v8f o0 = {}, o1 = {}, o2 = {}, o3 = {};
  const float sc = 0.051031036307982884f;  // C^-0.5 (note: n_embeds, not D)
  int col   = lane & 15;
  int rbase = (lane & 16) ? 8 : 0;

  for (int kt0 = 0; kt0 <= qt; kt0 += 2) {
    int kt1 = kt0 + 1;
    // --- score tiles S0 (keys kt0*16..+15) and S1 ---
    v8f s0 = {}, s1 = {};
    {
      const _Float16* kb = kb0 + (size_t)kt0 * 16 * Dn;
      s0 = wmma16(qa0, ldB16(kb, Dn), s0);
      s0 = wmma16(qa1, ldB16(kb + 32, Dn), s0);
    }
    if (kt1 <= qt) {  // wave-uniform branch: EXEC stays full for WMMA
      const _Float16* kb = kb0 + (size_t)kt1 * 16 * Dn;
      s1 = wmma16(qa0, ldB16(kb, Dn), s1);
      s1 = wmma16(qa1, ldB16(kb + 32, Dn), s1);
    }
    // --- scale + causal mask ---
#pragma unroll
    for (int r = 0; r < 8; r++) {
      int trow = qt * 16 + rbase + r;
      float a = s0[r] * sc, b = s1[r] * sc;
      if (kt0 * 16 + col > trow) a = -1e30f;
      if (kt1 > qt || kt1 * 16 + col > trow) b = -1e30f;
      s0[r] = a; s1[r] = b;
    }
    // --- online softmax (each row lives in one 16-lane half) ---
#pragma unroll
    for (int r = 0; r < 8; r++) {
      float tm = fmaxf(s0[r], s1[r]);
#pragma unroll
      for (int m = 1; m < 16; m <<= 1) tm = fmaxf(tm, __shfl_xor(tm, m, 32));
      float mn    = fmaxf(mrow[r], tm);
      float alpha = __expf(mrow[r] - mn);
      float p0    = __expf(s0[r] - mn);
      float p1    = __expf(s1[r] - mn);
      float ps    = p0 + p1;
#pragma unroll
      for (int m = 1; m < 16; m <<= 1) ps += __shfl_xor(ps, m, 32);
      lrow[r] = lrow[r] * alpha + ps;
      mrow[r] = mn;
      o0[r] *= alpha; o1[r] *= alpha; o2[r] *= alpha; o3[r] *= alpha;
      // C-frag -> row-major [16][32] f16 in LDS (A-frag source for P.V)
      myp[(rbase + r) * 32 + col]      = (_Float16)p0;
      myp[(rbase + r) * 32 + 16 + col] = (_Float16)p1;
    }
    // --- P(16x32) . V(32x64) via 4 WMMAs, P-frag reused across d tiles ---
    v16h pa = ldA16(myp, 32);
    const _Float16* vb = vb0 + kt0 * 16;
    o0 = wmma16(pa, ldB16(vb,           Tn), o0);
    o1 = wmma16(pa, ldB16(vb + 16 * Tn, Tn), o1);
    o2 = wmma16(pa, ldB16(vb + 32 * Tn, Tn), o2);
    o3 = wmma16(pa, ldB16(vb + 48 * Tn, Tn), o3);
  }
  // --- normalize and store concat-head output [b][t][h*64+d] as f16 ---
  int b = bh / Hn, h = bh - b * Hn;
#pragma unroll
  for (int r = 0; r < 8; r++) {
    float inv = 1.f / lrow[r];
    int t = qt * 16 + rbase + r;
    _Float16* op = out + ((size_t)b * Tn + t) * Cn + h * Dn;
    op[col]      = (_Float16)(o0[r] * inv);
    op[16 + col] = (_Float16)(o1[r] * inv);
    op[32 + col] = (_Float16)(o2[r] * inv);
    op[48 + col] = (_Float16)(o3[r] * inv);
  }
}

// ---- generic WMMA GEMM, 32x32 tile/wave:
//      C = A[MxK] * Bt[NxK]^T (+bias) (+resid) (relu?) -> f32 or f16 --------
__global__ void gemm_kernel(const _Float16* __restrict__ A,
                            const _Float16* __restrict__ Bt,
                            const float* __restrict__ bias,
                            const float* __restrict__ resid,
                            float* __restrict__ outF,
                            _Float16* __restrict__ outH,
                            int N, int K, int relu) {
  int n0 = blockIdx.x * 32;
  int m0 = (blockIdx.y * 4 + threadIdx.y) * 32;
  const _Float16* A0 = A + (size_t)m0 * K;
  const _Float16* A1 = A0 + (size_t)16 * K;
  const _Float16* B0 = Bt + (size_t)n0 * K;
  const _Float16* B1 = B0 + (size_t)16 * K;
  v8f c00 = {}, c01 = {}, c10 = {}, c11 = {};
  for (int kk = 0; kk < K; kk += 32) {
    v16h a0 = ldA16(A0 + kk, K), a1 = ldA16(A1 + kk, K);
    v16h b0 = ldB16(B0 + kk, K), b1 = ldB16(B1 + kk, K);
    c00 = wmma16(a0, b0, c00); c01 = wmma16(a0, b1, c01);
    c10 = wmma16(a1, b0, c10); c11 = wmma16(a1, b1, c11);
  }
  int lane = threadIdx.x;
  int nc = lane & 15;
  int rb = (lane & 16) ? 8 : 0;
#pragma unroll
  for (int mi = 0; mi < 2; mi++) {
#pragma unroll
    for (int ni = 0; ni < 2; ni++) {
      v8f acc = (mi == 0) ? (ni == 0 ? c00 : c01) : (ni == 0 ? c10 : c11);
      int n = n0 + ni * 16 + nc;
#pragma unroll
      for (int r = 0; r < 8; r++) {
        int m = m0 + mi * 16 + rb + r;
        float v = acc[r];
        if (bias)  v += bias[n];
        if (resid) v += resid[(size_t)m * N + n];
        if (relu)  v = fmaxf(v, 0.f);
        if (outF) outF[(size_t)m * N + n] = v;
        else      outH[(size_t)m * N + n] = (_Float16)v;
      }
    }
  }
}

// ---------------------------------------------------------------------------
extern "C" void kernel_launch(void* const* d_in, const int* in_sizes, int n_in,
                              void* d_out, int out_size, void* d_ws, size_t ws_size,
                              hipStream_t stream) {
  (void)in_sizes; (void)n_in; (void)out_size; (void)ws_size;
  const float* x      = (const float*)d_in[0];
  const float* wq     = (const float*)d_in[1];
  const float* wk     = (const float*)d_in[2];
  const float* wv     = (const float*)d_in[3];
  const float* w_proj = (const float*)d_in[4];
  const float* b_proj = (const float*)d_in[5];
  const float* w1     = (const float*)d_in[6];
  const float* b1     = (const float*)d_in[7];
  const float* w2     = (const float*)d_in[8];
  const float* b2     = (const float*)d_in[9];
  const float* g1     = (const float*)d_in[10];
  const float* be1    = (const float*)d_in[11];
  const float* g2     = (const float*)d_in[12];
  const float* be2    = (const float*)d_in[13];

  // workspace carve-up (~79 MB total), 256B aligned slices
  char* p = (char*)d_ws;
  auto take = [&](size_t bytes) {
    char* r = p; p += (bytes + 255) & ~(size_t)255; return r;
  };
  _Float16* wqkvT  = (_Float16*)take((size_t)3 * Hn * Dn * Cn * 2); // [3][H][D][C]
  _Float16* wpT    = (_Float16*)take((size_t)Cn * Cn * 2);          // [C][C]
  _Float16* w1T    = (_Float16*)take((size_t)Fn * Cn * 2);          // [4C][C]
  _Float16* w2T    = (_Float16*)take((size_t)Cn * Fn * 2);          // [C][4C]
  _Float16* h1     = (_Float16*)take((size_t)Mn * Cn * 2);
  _Float16* qbuf   = (_Float16*)take((size_t)Bn * Hn * Tn * Dn * 2);
  _Float16* kbuf   = (_Float16*)take((size_t)Bn * Hn * Tn * Dn * 2);
  _Float16* vTbuf  = (_Float16*)take((size_t)Bn * Hn * Dn * Tn * 2);
  _Float16* attn_o = (_Float16*)take((size_t)Mn * Cn * 2);
  float*    x_sa   = (float*)   take((size_t)Mn * Cn * 4);
  _Float16* h2     = (_Float16*)take((size_t)Mn * Cn * 2);
  _Float16* ff1    = (_Float16*)take((size_t)Mn * Fn * 2);

  dim3 wv4(32, 4);
  auto nb = [](size_t n) { return (unsigned)((n + 255) / 256); };

  // 1) weight transpose + cast to f16 [N][K]
  transpose_cvt_kernel<<<nb((size_t)Hn*Cn*Dn), 256, 0, stream>>>(wq, wqkvT,                      Hn, Cn, Dn);
  transpose_cvt_kernel<<<nb((size_t)Hn*Cn*Dn), 256, 0, stream>>>(wk, wqkvT + (size_t)Hn*Dn*Cn,   Hn, Cn, Dn);
  transpose_cvt_kernel<<<nb((size_t)Hn*Cn*Dn), 256, 0, stream>>>(wv, wqkvT + (size_t)2*Hn*Dn*Cn, Hn, Cn, Dn);
  transpose_cvt_kernel<<<nb((size_t)Cn*Cn),    256, 0, stream>>>(w_proj, wpT, 1, Cn, Cn);
  transpose_cvt_kernel<<<nb((size_t)Cn*Fn),    256, 0, stream>>>(w1, w1T, 1, Cn, Fn);
  transpose_cvt_kernel<<<nb((size_t)Fn*Cn),    256, 0, stream>>>(w2, w2T, 1, Fn, Cn);

  // 2) LN1
  layernorm_kernel<<<dim3(Mn / 4), wv4, 0, stream>>>(x, g1, be1, h1);
  // 3) QKV projections (WMMA, 32x32 tiles), V transposed
  qkv_kernel<<<dim3((3 * Hn * Dn) / 32, Mn / 128), wv4, 0, stream>>>(h1, wqkvT, qbuf, kbuf, vTbuf);
  // 4) flash attention (WMMA scores + WMMA P.V)
  attn_kernel<<<dim3(Tn / 64, Bn * Hn), wv4, 0, stream>>>(qbuf, kbuf, vTbuf, attn_o);
  // 5) output projection + bias + residual(x) -> x_sa (fp32)
  gemm_kernel<<<dim3(Cn / 32, Mn / 128), wv4, 0, stream>>>(attn_o, wpT, b_proj, x, x_sa, nullptr, Cn, Cn, 0);
  // 6) LN2
  layernorm_kernel<<<dim3(Mn / 4), wv4, 0, stream>>>(x_sa, g2, be2, h2);
  // 7) FF1 + bias + ReLU -> f16
  gemm_kernel<<<dim3(Fn / 32, Mn / 128), wv4, 0, stream>>>(h2, w1T, b1, nullptr, nullptr, ff1, Fn, Cn, 1);
  // 8) FF2 + bias + residual(x_sa) -> d_out (fp32)
  gemm_kernel<<<dim3(Cn / 32, Mn / 128), wv4, 0, stream>>>(ff1, w2T, b2, x_sa, (float*)d_out, nullptr, Cn, Fn, 0);
}
